// NetVLAD_75788992905756
// MI455X (gfx1250) — matrix operand order repaired and checked by reading, not compiled, not run on hardware
//
#include <hip/hip_runtime.h>
#include <hip/hip_bf16.h>

// NetVLAD fused kernel for gfx1250 (MI455X), wave32 WMMA bf16 path.
// Stage A: logits = X@W via v_wmma_f32_16x16x32_bf16 (f32 logits kept in LDS),
//          softmax in f32 -> A (bf16, transposed in LDS) + Asum via ds_add_f32.
// Stage B: S = X^T @ A via WMMA, epilogue V = S - C*Asum written as [B,K,D].

typedef __attribute__((ext_vector_type(16))) __bf16 v16bf;
typedef __attribute__((ext_vector_type(8)))  __bf16 v8bf;
typedef __attribute__((ext_vector_type(8)))  float  v8f;

#define NPIX 196   // 14*14
#define NPAD 224   // 7*32 padded pixel count
#define DD   512
#define KK   64
#define DC   32    // stage-A D-chunk width

// Build a v16bf A/B fragment from two aligned 16-byte LDS chunks.
__device__ __forceinline__ v16bf ld_frag16(const __bf16* lo, const __bf16* hi) {
  v8bf l = *(const v8bf*)lo;
  v8bf h = *(const v8bf*)hi;
  v16bf r;
#pragma unroll
  for (int i = 0; i < 8; ++i) { r[i] = l[i]; r[i + 8] = h[i]; }
  return r;
}

struct SA { __bf16 X[NPAD][DC]; __bf16 Wt[KK][DC]; };   // 18,432 B
struct SB { __bf16 Xt[DD][32]; };                        // 32,768 B
union  SU {
  SA a;                       // stage-A staging
  SB b;                       // stage-B staging
  float L[KK][NPAD];          // f32 logits (transposed), 57,344 B — live only between stages
};

__global__ __launch_bounds__(256) void netvlad_fused_kernel(
    const float* __restrict__ X, const float* __restrict__ Wm,
    const float* __restrict__ Cc, float* __restrict__ Out) {
  __shared__ SU u;                       // 57,344 B
  __shared__ __bf16 AsT[KK][NPAD];       // 28,672 B  softmax A, transposed [k][pixel]
  __shared__ float  Asum[KK];            //    256 B

  const int b    = blockIdx.x;           // one workgroup per batch image
  const int tid  = threadIdx.x;          // 256 threads = 8 waves
  const int lane = tid & 31;
  const int wv   = tid >> 5;             // wave id 0..7
  const int m    = lane & 15;            // fragment row/col within 16
  const int g    = lane >> 4;            // lane half-group

  const float* Xb = X + (size_t)b * NPIX * DD;

  if (tid < KK) Asum[tid] = 0.f;

  // ---------------- Stage A: logits tiles 14(M=pixels) x 4(N=clusters) ----------------
  const v8f vzero = {};
  v8f accA[7];
#pragma unroll
  for (int i = 0; i < 7; ++i) accA[i] = vzero;

  for (int c = 0; c < DD / DC; ++c) {
    __syncthreads();
    // X chunk [NPAD x DC] as bf16, padding rows zeroed
    for (int idx = tid; idx < NPAD * DC; idx += 256) {
      int p = idx / DC, j = idx % DC;
      float v = (p < NPIX) ? Xb[p * DD + c * DC + j] : 0.f;
      u.a.X[p][j] = (__bf16)v;
      // warm L2->WGP path for next chunk while this chunk computes
      if (c + 1 < DD / DC && p < NPIX && j == 0)
        __builtin_prefetch(&Xb[p * DD + (c + 1) * DC], 0, 1);
    }
    // W chunk transposed: Wt[k][j] = W[(c*DC+j), k]
    for (int idx = tid; idx < KK * DC; idx += 256) {
      int k = idx / DC, j = idx % DC;
      u.a.Wt[k][j] = (__bf16)Wm[(c * DC + j) * KK + k];
    }
    __syncthreads();
#pragma unroll
    for (int i = 0; i < 7; ++i) {            // 56 tiles, 7 per wave
      int t  = wv * 7 + i;
      int mt = t >> 2, nt = t & 3;
      v16bf a  = ld_frag16(&u.a.X[mt * 16 + m][g * 8],
                           &u.a.X[mt * 16 + m][16 + g * 8]);
      v16bf bb = ld_frag16(&u.a.Wt[nt * 16 + m][g * 8],
                           &u.a.Wt[nt * 16 + m][16 + g * 8]);
      accA[i] = __builtin_amdgcn_wmma_f32_16x16x32_bf16(
          false, a, false, bb, (short)0, accA[i], false, false);
    }
  }
  __syncthreads();
  // spill logit fragments (f32) transposed into u.L[k][p]  (union region now free)
#pragma unroll
  for (int i = 0; i < 7; ++i) {
    int t  = wv * 7 + i;
    int mt = t >> 2, nt = t & 3;
    int k  = nt * 16 + m;
#pragma unroll
    for (int r = 0; r < 8; ++r) {
      int p = mt * 16 + r + 8 * g;
      u.L[k][p] = accA[i][r];
    }
  }
  __syncthreads();

  // ---------------- softmax over K=64 clusters (f32), one pixel-row per thread ---------
  if (tid < NPAD) {
    if (tid >= NPIX) {
      for (int k = 0; k < KK; ++k) AsT[k][tid] = (__bf16)0.f;   // kill padding pixels
    } else {
      float mx = -1e30f;
#pragma unroll
      for (int k = 0; k < KK; ++k) mx = fmaxf(mx, u.L[k][tid]);
      float s = 0.f;
      float e[KK];
#pragma unroll
      for (int k = 0; k < KK; ++k) { e[k] = __expf(u.L[k][tid] - mx); s += e[k]; }
      float inv = 1.f / s;
#pragma unroll
      for (int k = 0; k < KK; ++k) {
        float a = e[k] * inv;
        AsT[k][tid] = (__bf16)a;
        atomicAdd(&Asum[k], a);            // ds_add_f32
      }
    }
  }
  __syncthreads();

  // ---------------- Stage B: S = X^T @ A, tiles 32(M=D) x 4(N=K) --------------------
  const size_t baseOut = (size_t)b * KK * DD;
  for (int hh = 0; hh < 2; ++hh) {         // two N-tile halves to cap VGPR pressure
    v8f accB[4][2];
#pragma unroll
    for (int mi = 0; mi < 4; ++mi)
#pragma unroll
      for (int j = 0; j < 2; ++j) accB[mi][j] = vzero;

    for (int pc = 0; pc < NPAD / 32; ++pc) {   // 7 K-steps of 32 pixels
      __syncthreads();
      // X^T chunk: Xt[d][q] = X[b][pc*32+q][d] (coalesced reads, transposed LDS store)
      for (int idx = tid; idx < DD * 32; idx += 256) {
        int q = idx / DD, d = idx % DD;
        int p = pc * 32 + q;
        float v = (p < NPIX) ? Xb[p * DD + d] : 0.f;
        u.b.Xt[d][q] = (__bf16)v;
        if (pc + 1 < NPAD / 32 && (p + 32) < NPIX && (d & 31) == 0)
          __builtin_prefetch(&Xb[(p + 32) * DD + d], 0, 1);   // global_prefetch_b8
      }
      __syncthreads();
#pragma unroll
      for (int mi = 0; mi < 4; ++mi) {
        int mt = wv + 8 * mi;
        v16bf a = ld_frag16(&u.b.Xt[mt * 16 + m][g * 8],
                            &u.b.Xt[mt * 16 + m][16 + g * 8]);
#pragma unroll
        for (int j = 0; j < 2; ++j) {
          int nt = hh * 2 + j;
          v16bf bb = ld_frag16(&AsT[nt * 16 + m][pc * 32 + g * 8],
                               &AsT[nt * 16 + m][pc * 32 + 16 + g * 8]);
          accB[mi][j] = __builtin_amdgcn_wmma_f32_16x16x32_bf16(
              false, a, false, bb, (short)0, accB[mi][j], false, false);
        }
      }
    }
    // epilogue: V[d,k] = S - C[d,k]*Asum[k], write as Out[b][k][d]
#pragma unroll
    for (int mi = 0; mi < 4; ++mi) {
      int mt = wv + 8 * mi;
#pragma unroll
      for (int j = 0; j < 2; ++j) {
        int nt = hh * 2 + j;
        int k  = nt * 16 + m;
        float ak = Asum[k];
#pragma unroll
        for (int r = 0; r < 8; ++r) {
          int d = mt * 16 + r + 8 * g;
          Out[baseOut + (size_t)k * DD + d] = accB[mi][j][r] - Cc[d * KK + k] * ak;
        }
      }
    }
  }
}

extern "C" void kernel_launch(void* const* d_in, const int* in_sizes, int n_in,
                              void* d_out, int out_size, void* d_ws, size_t ws_size,
                              hipStream_t stream) {
  (void)in_sizes; (void)n_in; (void)d_ws; (void)ws_size; (void)out_size;
  const float* X = (const float*)d_in[0];   // [32,14,14,512]
  const float* W = (const float*)d_in[1];   // [512,64]
  const float* C = (const float*)d_in[2];   // [1,1,1,512,64]
  float* Out = (float*)d_out;               // [32,64,512]
  netvlad_fused_kernel<<<dim3(32), dim3(256), 0, stream>>>(X, W, C, Out);
}
